// TemporalSlotAttention_87273735455257
// MI455X (gfx1250) — compile-verified, matrix operand order checked by reading, not run on hardware
//
#include <hip/hip_runtime.h>
#include <hip/hip_fp16.h>

typedef __attribute__((ext_vector_type(16))) _Float16 v16h;
typedef __attribute__((ext_vector_type(8)))  _Float16 v8h;
typedef __attribute__((ext_vector_type(4)))  _Float16 v4h;
typedef __attribute__((ext_vector_type(8)))  float    v8f;
typedef __attribute__((ext_vector_type(4)))  float    v4f;
typedef __attribute__((ext_vector_type(4)))  unsigned int v4u;
typedef __attribute__((ext_vector_type(8)))  int      v8i_;
typedef __attribute__((ext_vector_type(4)))  int      v4i_;

#define BB     32
#define TT     16
#define NSLOT  64
#define DD     512
#define ROWS   (BB*TT*NSLOT)      /* 32768 rows of [*, D] */
#define LN_EPS 1e-5f
#define EPS_R  1e-8f
#define SCALE  0.044194173824159216f   /* 512^-0.5 */

// LDS weight-tile row stride in halves: 256 data + 8 pad (TDM pad: 16B per row)
#define WT_STRIDE 264

// Load one WMMA 16-bit fragment (A-row or B-column) for 16x16x32:
// lane handles row/col (lane&15); K chunks are [cb..cb+7] and [cb+16..cb+23]
// with cb=(lane>>4)*8 already folded into `base` by the caller.
__device__ __forceinline__ v16h load_frag16(const _Float16* base) {
  v8h lo = *(const v8h*)(base);
  v8h hi = *(const v8h*)(base + 16);
  return __builtin_shufflevector(lo, hi, 0,1,2,3,4,5,6,7,8,9,10,11,12,13,14,15);
}

__device__ __forceinline__ v8f wmma_f16(v16h a, v16h b, v8f c) {
  return __builtin_amdgcn_wmma_f32_16x16x32_f16(false, a, false, b, (short)0, c,
                                                false, false);
}

// ---------------------------------------------------------------------------
// Kernel 1: LayerNorm over D=512, write f16 norm. One wave per row.
// ---------------------------------------------------------------------------
__global__ __launch_bounds__(256) void ln_kernel(const float* __restrict__ x,
                                                 const float* __restrict__ g,
                                                 const float* __restrict__ be,
                                                 _Float16* __restrict__ normH) {
  int wave = threadIdx.x >> 5;
  int lane = threadIdx.x & 31;
  int row  = blockIdx.x * 8 + wave;
  const float* xr = x + (size_t)row * DD;

  float vals[16];
  float s = 0.f;
#pragma unroll
  for (int c = 0; c < 4; c++) {
    v4f f = *(const v4f*)(xr + c * 128 + lane * 4);
#pragma unroll
    for (int k = 0; k < 4; k++) { vals[c * 4 + k] = f[k]; s += f[k]; }
  }
#pragma unroll
  for (int off = 16; off >= 1; off >>= 1) s += __shfl_xor(s, off, 32);
  float mu = s * (1.f / DD);

  float vs = 0.f;
#pragma unroll
  for (int k = 0; k < 16; k++) { float d = vals[k] - mu; vs += d * d; }
#pragma unroll
  for (int off = 16; off >= 1; off >>= 1) vs += __shfl_xor(vs, off, 32);
  float rs = rsqrtf(vs * (1.f / DD) + LN_EPS);

  _Float16* orow = normH + (size_t)row * DD;
#pragma unroll
  for (int c = 0; c < 4; c++) {
    v4h h;
#pragma unroll
    for (int k = 0; k < 4; k++) {
      int col = c * 128 + lane * 4 + k;
      h[k] = (_Float16)((vals[c * 4 + k] - mu) * rs * g[col] + be[col]);
    }
    *(v4h*)(orow + c * 128 + lane * 4) = h;
  }
}

// ---------------------------------------------------------------------------
// Kernel 2: convert Wq|Wk|Wv (f32) -> concatenated WH (f16), 3*512*512 elems
// ---------------------------------------------------------------------------
__global__ __launch_bounds__(256) void wcvt_kernel(const float* __restrict__ wq,
                                                   const float* __restrict__ wk,
                                                   const float* __restrict__ wv,
                                                   _Float16* __restrict__ WH) {
  int i = blockIdx.x * 256 + threadIdx.x;           // 0 .. 786431
  const float* src = (i < 262144) ? wq : ((i < 524288) ? wk : wv);
  WH[i] = (_Float16)src[i & 262143];
}

// ---------------------------------------------------------------------------
// Kernel 3: QKV GEMM.  [32768 x 512] x [512 x 1536]^T via WMMA f16.
// Per block: 128(M) x 64(N); per wave: 16(M) x 64(N) = 4 accumulators.
// The 64x512 weight tile is DMA'd into LDS by the Tensor Data Mover in two
// 64x256 halves (TDM row padding of 16B -> conflict-free ds_load_b128 reads).
// Q,K stored row-major f16; V stored TRANSPOSED: Vt[b][d][t*64+n] f16.
// ---------------------------------------------------------------------------
__global__ __launch_bounds__(256) void qkv_gemm_kernel(
    const _Float16* __restrict__ A, const _Float16* __restrict__ WH,
    const float* __restrict__ bq, const float* __restrict__ bk,
    const float* __restrict__ bv,
    _Float16* __restrict__ QH, _Float16* __restrict__ KH,
    _Float16* __restrict__ VtH) {
  __shared__ _Float16 wtile[64 * WT_STRIDE];        // 33,792 B

  int wave = threadIdx.x >> 5, lane = threadIdx.x & 31;
  int mb = blockIdx.x & 255;        // 256 M-blocks of 128
  int nb = blockIdx.x >> 8;         // 24 N-blocks of 64 (over 1536 features)
  int m0 = mb * 128 + wave * 16;
  int n0 = nb * 64;
  int r  = lane & 15;
  int cb = (lane >> 4) * 8;

  v8f acc[4] = {v8f{}, v8f{}, v8f{}, v8f{}};
  const _Float16* aptr = A + (size_t)(m0 + r) * DD + cb;

  for (int half = 0; half < 2; half++) {
    int kbase = half * 256;
    // ---- TDM: DMA 64 rows x 256 halves of WH into LDS (wave 0 issues) ----
    if (threadIdx.x < 32) {
      unsigned long long ga =
          (unsigned long long)(uintptr_t)(WH + (size_t)n0 * DD + kbase);
      unsigned int lds = (unsigned int)(uintptr_t)wtile;  // low 32 = LDS offset
      v4u g0;
      g0[0] = 1u;                                   // count=1 valid descriptor
      g0[1] = lds;                                  // lds_addr
      g0[2] = (unsigned int)ga;                     // global_addr[31:0]
      g0[3] = (unsigned int)(ga >> 32) | (2u << 30);// global_addr[56:32],type=2
      v8i_ g1;
      // data_size=2B, pad_enable, pad_interval=6 (128 dwords), pad_amount=3 (4 dwords)
      g1[0] = (int)0x07910000u;
      g1[1] = (int)(512u << 16);   // tensor_dim0 = 512 (low16 at bits 63:48->[31:16])
      g1[2] = (int)(64u << 16);    // tensor_dim0 hi=0 | tensor_dim1 = 64
      g1[3] = (int)(256u << 16);   // tensor_dim1 hi=0 | tile_dim0 = 256
      g1[4] = 64;                  // tile_dim1 = 64, tile_dim2 = 0
      g1[5] = 512;                 // tensor_dim0_stride low 32
      g1[6] = 0;                   // stride hi | tensor_dim1_stride = 0
      g1[7] = 0;
      v4i_ z4 = {};
      v8i_ z8 = {};
      __builtin_amdgcn_tensor_load_to_lds(g0, g1, z4, z4, z8, 0);
      __builtin_amdgcn_s_wait_tensorcnt(0);
    }
    __syncthreads();

    // ---- 8 K-steps of 32 against the staged weight tile ----
#pragma unroll 2
    for (int kk = 0; kk < 256; kk += 32) {
      __builtin_prefetch(aptr + kbase + kk + 128, 0, 1);  // global_prefetch_b8
      v16h afrag = load_frag16(aptr + kbase + kk);
#pragma unroll
      for (int s = 0; s < 4; s++) {
        const _Float16* bp = wtile + (s * 16 + r) * WT_STRIDE + kk + cb;
        v16h bfrag = load_frag16(bp);               // ds_load_b128 x2
        acc[s] = wmma_f16(afrag, bfrag, acc[s]);
      }
    }
    __syncthreads();   // protect wtile before next TDM overwrite
  }

#pragma unroll
  for (int s = 0; s < 4; s++) {
    int ng    = n0 + s * 16 + (lane & 15);          // 0..1535 output feature
    int mat   = ng >> 9;                            // 0=q 1=k 2=v
    int col   = ng & 511;
    int mbase = m0 + ((lane >> 4) * 8);
    float bia = (mat == 0 ? bq : (mat == 1 ? bk : bv))[col];
    if (mat == 0) {
#pragma unroll
      for (int rr = 0; rr < 8; rr++)
        QH[(size_t)(mbase + rr) * DD + col] = (_Float16)(acc[s][rr] + bia);
    } else if (mat == 1) {
#pragma unroll
      for (int rr = 0; rr < 8; rr++)
        KH[(size_t)(mbase + rr) * DD + col] = (_Float16)(acc[s][rr] + bia);
    } else {
#pragma unroll
      for (int rr = 0; rr < 8; rr++) {
        int m  = mbase + rr;
        int bb = m >> 10;                           // batch
        int w  = m & 1023;                          // t*64 + slot
        VtH[((size_t)bb * DD + col) * 1024 + w] = (_Float16)(acc[s][rr] + bia);
      }
    }
  }
}

// ---------------------------------------------------------------------------
// Kernel 4: dots = Q.K^T over (t,d)=8192, scale, softmax over query axis,
// eps renorm over keys. One block (16 waves) per batch; wave = one 16x16 tile.
// ---------------------------------------------------------------------------
__global__ __launch_bounds__(512) void attn_kernel(const _Float16* __restrict__ QH,
                                                   const _Float16* __restrict__ KH,
                                                   _Float16* __restrict__ attnH) {
  __shared__ float S[64][65];
  int b    = blockIdx.x;
  int wave = threadIdx.x >> 5, lane = threadIdx.x & 31;
  int i0 = (wave >> 2) * 16, j0 = (wave & 3) * 16;
  int r  = lane & 15;
  int cb = (lane >> 4) * 8;

  v8f acc = {};
  const _Float16* qb = QH + (size_t)b * 1024 * DD;
  const _Float16* kb = KH + (size_t)b * 1024 * DD;
  for (int t = 0; t < TT; t++) {
    const _Float16* qt = qb + (size_t)(t * 64 + i0 + r) * DD + cb;
    const _Float16* kt = kb + (size_t)(t * 64 + j0 + r) * DD + cb;
#pragma unroll 4
    for (int kk = 0; kk < DD; kk += 32) {
      v16h a = load_frag16(qt + kk);
      v16h k = load_frag16(kt + kk);
      acc = wmma_f16(a, k, acc);
    }
  }
#pragma unroll
  for (int rr = 0; rr < 8; rr++)
    S[i0 + rr + (lane >> 4) * 8][j0 + (lane & 15)] = acc[rr] * SCALE;
  __syncthreads();

  // softmax over query axis i (per key-column j), then + eps
  if (threadIdx.x < 64) {
    int j = threadIdx.x;
    float m = -INFINITY;
    for (int i = 0; i < 64; i++) m = fmaxf(m, S[i][j]);
    float sum = 0.f;
    for (int i = 0; i < 64; i++) sum += __expf(S[i][j] - m);
    float inv = 1.f / sum;
    for (int i = 0; i < 64; i++) S[i][j] = __expf(S[i][j] - m) * inv + EPS_R;
  }
  __syncthreads();
  // renormalize over keys j (per query-row i), write f16 attention
  if (threadIdx.x < 64) {
    int i = threadIdx.x;
    float sum = 0.f;
    for (int j = 0; j < 64; j++) sum += S[i][j];
    float inv = 1.f / sum;
    for (int j = 0; j < 64; j++)
      attnH[((size_t)b * 64 + i) * 64 + j] = (_Float16)(S[i][j] * inv);
  }
}

// ---------------------------------------------------------------------------
// Kernel 5: updates[b,i,t*512+d] = sum_j attn[b,i,j] * V[b,t*64+j,d]
// per (b,t): [64x64] x [64x512] GEMM via WMMA (K=64 -> 2 steps of 32).
// Output written directly in [b,t,n,d] order (transpose folded into store).
// ---------------------------------------------------------------------------
__global__ __launch_bounds__(512) void update_kernel(const _Float16* __restrict__ attnH,
                                                     const _Float16* __restrict__ VtH,
                                                     float* __restrict__ out) {
  int bx = blockIdx.x;
  int b = bx >> 4, t = bx & 15;
  int wave = threadIdx.x >> 5, lane = threadIdx.x & 31;
  int it = wave & 3, dg = wave >> 2;   // 4 i-tiles x 4 d-groups
  int i0 = it * 16;
  int r  = lane & 15;
  int cb = (lane >> 4) * 8;

  const _Float16* ap = attnH + ((size_t)b * 64 + i0 + r) * 64 + cb;
  v16h a0 = load_frag16(ap);        // j = 0..31
  v16h a1 = load_frag16(ap + 32);   // j = 32..63

#pragma unroll 2
  for (int s = 0; s < 8; s++) {
    int d0 = (dg * 8 + s) * 16;
    const _Float16* vp = VtH + ((size_t)b * DD + d0 + r) * 1024 + t * 64 + cb;
    v16h b0 = load_frag16(vp);
    v16h b1 = load_frag16(vp + 32);
    v8f acc = {};
    acc = wmma_f16(a0, b0, acc);
    acc = wmma_f16(a1, b1, acc);
    float* op = out + ((size_t)((b * 16 + t) * 64) + i0 + (lane >> 4) * 8) * DD
                + d0 + (lane & 15);
#pragma unroll
    for (int rr = 0; rr < 8; rr++) op[(size_t)rr * DD] = acc[rr];
  }
}

// ---------------------------------------------------------------------------
extern "C" void kernel_launch(void* const* d_in, const int* in_sizes, int n_in,
                              void* d_out, int out_size, void* d_ws, size_t ws_size,
                              hipStream_t stream) {
  (void)in_sizes; (void)n_in; (void)out_size; (void)ws_size;
  const float* in_slots = (const float*)d_in[0];
  const float* Wq = (const float*)d_in[1];
  const float* bq = (const float*)d_in[2];
  const float* Wk = (const float*)d_in[3];
  const float* bk = (const float*)d_in[4];
  const float* Wv = (const float*)d_in[5];
  const float* bv = (const float*)d_in[6];
  const float* ln_g = (const float*)d_in[7];
  const float* ln_b = (const float*)d_in[8];
  float* out = (float*)d_out;

  char* ws = (char*)d_ws;
  const size_t SZ_NORM = (size_t)ROWS * DD * 2;           // 32 MiB
  const size_t SZ_W    = (size_t)3 * DD * DD * 2;         // 1.5 MiB
  _Float16* normH = (_Float16*)(ws);
  _Float16* WH    = (_Float16*)(ws + SZ_NORM);
  _Float16* QH    = (_Float16*)(ws + SZ_NORM + SZ_W);
  _Float16* KH    = (_Float16*)(ws + 2 * SZ_NORM + SZ_W);
  _Float16* VtH   = (_Float16*)(ws + 3 * SZ_NORM + SZ_W);
  _Float16* attnH = (_Float16*)(ws + 4 * SZ_NORM + SZ_W);

  ln_kernel<<<ROWS / 8, 256, 0, stream>>>(in_slots, ln_g, ln_b, normH);
  wcvt_kernel<<<(3 * DD * DD) / 256, 256, 0, stream>>>(Wq, Wk, Wv, WH);
  qkv_gemm_kernel<<<256 * 24, 256, 0, stream>>>(normH, WH, bq, bk, bv,
                                                QH, KH, VtH);
  attn_kernel<<<BB, 512, 0, stream>>>(QH, KH, attnH);
  update_kernel<<<BB * TT, 512, 0, stream>>>(attnH, VtH, out);
}
